// NeRF_7421703488253
// MI455X (gfx1250) — matrix operand order repaired and checked by reading, not compiled, not run on hardware
//
#include <hip/hip_runtime.h>
#include <hip/hip_bf16.h>

// ---------------------------------------------------------------------------
// NeRF forward on MI455X (gfx1250): f16 WMMA for the 256x256 and 256x128
// GEMMs, VALU for the tiny heads, wave-scan compositing.
// One block = one ray (128 samples); software-pipelined fragment loads.
// ---------------------------------------------------------------------------

typedef __attribute__((ext_vector_type(16))) _Float16 v16h;
typedef __attribute__((ext_vector_type(8)))  _Float16 v8h;
typedef __attribute__((ext_vector_type(8)))  float    v8f;

static constexpr int H   = 256;   // hidden width
static constexpr int HC  = 128;   // color hidden width
static constexpr int MT  = 128;   // points (rows) per block = one ray
static constexpr int LDA = 264;   // LDS row stride in halves (16B-mult, bank skew)

// fragment combine: two 16-byte chunks -> v16h
__device__ __forceinline__ v16h frag_cat(const _Float16* lo, const _Float16* hi) {
    v8h a = *(const v8h*)lo;
    v8h b = *(const v8h*)hi;
    return __builtin_shufflevector(a, b, 0,1,2,3,4,5,6,7,8,9,10,11,12,13,14,15);
}

// ---------------------------------------------------------------------------
// Pack W2 (256x256) and W3[0:256] (256x128) f32 -> f16 fragment-major tiles.
// A 32x16 (KxN) block is stored as 32 lanes x 16 contiguous halves:
//   lane L -> column n = nt*16 + (L&15), K-run base = kt*32 + (L>=16 ? 16 : 0)
// ---------------------------------------------------------------------------
__global__ __launch_bounds__(256) void nerf_pack_weights(
        const float* __restrict__ W2, const float* __restrict__ W3,
        _Float16* __restrict__ w2p, _Float16* __restrict__ w3p) {
    const int idx = blockIdx.x * 256 + threadIdx.x;
    const int TOT2 = (H / 32) * (H / 16) * 32 * 16;    // 65536
    const int TOT3 = (H / 32) * (HC / 16) * 32 * 16;   // 32768
    if (idx < TOT2) {
        int h = idx & 15, L = (idx >> 4) & 31, tile = idx >> 9;
        int nt = tile % (H / 16), kt = tile / (H / 16);
        int k = kt * 32 + ((L >> 4) << 4) + h;
        int n = nt * 16 + (L & 15);
        w2p[idx] = (_Float16)W2[k * H + n];
    } else if (idx < TOT2 + TOT3) {
        int i2 = idx - TOT2;
        int h = i2 & 15, L = (i2 >> 4) & 31, tile = i2 >> 9;
        int nt = tile % (HC / 16), kt = tile / (HC / 16);
        int k = kt * 32 + ((L >> 4) << 4) + h;
        int n = nt * 16 + (L & 15);
        w3p[i2] = (_Float16)W3[k * HC + n];
    }
}

// ---------------------------------------------------------------------------
// Fused MLP: one block (8 waves, 256 threads) = one ray = 128 samples.
// rgbd[p] = {r, g, b, density}
// ---------------------------------------------------------------------------
__global__ __launch_bounds__(256) void nerf_mlp(
        const float* __restrict__ x,  const float* __restrict__ d,
        const float* __restrict__ tv,
        const float* __restrict__ W1, const float* __restrict__ b1,
        const float* __restrict__ b2,
        const float* __restrict__ Wd, const float* __restrict__ bd,
        const float* __restrict__ W3, const float* __restrict__ b3,
        const float* __restrict__ W4, const float* __restrict__ b4,
        const _Float16* __restrict__ w2p, const _Float16* __restrict__ w3p,
        float* __restrict__ rgbd, int Nsamp) {
    __shared__ __align__(16) _Float16 act[MT][LDA];   // ~67.6 KB (gfx1250: 320KB/WGP)
    __shared__ float dcol[HC];
    __shared__ float tsh[MT];

    const int tid  = threadIdx.x;
    const int lane = tid & 31;
    const int wave = tid >> 5;
    const int p0   = blockIdx.x * MT;          // first point of block
    const int ray  = p0 / Nsamp;
    const int s0   = p0 % Nsamp;               // 0 when Nsamp == 128

    const float x0 = x[ray * 3 + 0], x1 = x[ray * 3 + 1], x2 = x[ray * 3 + 2];
    const float d0 = d[ray * 3 + 0], d1 = d[ray * 3 + 1], d2 = d[ray * 3 + 2];

    if (tid < MT) tsh[tid] = tv[s0 + tid];
    // per-column bias for layer 3: b3 + dir @ W3[256:259,:]  (dir is block-uniform)
    if (tid < HC) {
        dcol[tid] = b3[tid] + d0 * W3[(H + 0) * HC + tid]
                            + d1 * W3[(H + 1) * HC + tid]
                            + d2 * W3[(H + 2) * HC + tid];
    }
    __syncthreads();

    // ---- layer 1: h1 = relu(pts @ W1 + b1), K=3 (VALU) -------------------
    {
        const int c = tid;                      // one column per thread
        const float w0 = W1[c], w1 = W1[H + c], w2 = W1[2 * H + c], bb = b1[c];
        #pragma unroll 4
        for (int r = 0; r < MT; ++r) {
            const float t = tsh[r];
            float v = fmaf(fmaf(d0, t, x0), w0,
                      fmaf(fmaf(d1, t, x1), w1,
                      fmaf(fmaf(d2, t, x2), w2, bb)));
            act[r][c] = (_Float16)fmaxf(v, 0.0f);
        }
    }
    __syncthreads();

    // common fragment geometry for this wave: wave owns M-tile `wave`
    const int mt   = wave;
    const int koff = (lane >> 4) << 4;          // A-frag half-offset (0 / 16)
    const int mofs = (lane >> 4) << 3;          // D-frag row offset  (0 / 8)
    const _Float16* abase = &act[mt * 16 + (lane & 15)][0];

    // ---- layer 2: h2 = relu(h1 @ W2 + b2), 128x256, K=256 (WMMA) ---------
    {
        const _Float16* wbase = w2p + (size_t)lane * 16;   // + tile*512 halves
        v8f acc[16];
        #pragma unroll
        for (int j = 0; j < 16; ++j) acc[j] = (v8f)(0.0f);

        // software pipeline: prefetch A for kt+1 and B for tile j+1
        v16h afrag = frag_cat(abase + koff, abase + koff + 8);
        v16h bnext = frag_cat(wbase, wbase + 8);
        for (int kt = 0; kt < H / 32; ++kt) {
            const v16h acur = afrag;
            if (kt < H / 32 - 1) {
                const _Float16* ap = abase + (kt + 1) * 32 + koff;
                afrag = frag_cat(ap, ap + 8);
            }
            #pragma unroll
            for (int j = 0; j < 16; ++j) {
                const v16h bcur = bnext;
                // always prefetch next tile (over-reads 1KB past w2p into ws; unused)
                const _Float16* wp = wbase + (size_t)(kt * 16 + j + 1) * 512;
                bnext = frag_cat(wp, wp + 8);
                acc[j] = __builtin_amdgcn_wmma_f32_16x16x32_f16(
                    false, acur, false, bcur, (short)0, acc[j], false, false);
            }
        }
        __syncthreads();                        // all h1 reads complete
        #pragma unroll
        for (int j = 0; j < 16; ++j) {
            const int c = j * 16 + (lane & 15);
            const float bias = b2[c];
            #pragma unroll
            for (int v = 0; v < 8; ++v) {
                const int m = mt * 16 + v + mofs;
                act[m][c] = (_Float16)fmaxf(acc[j][v] + bias, 0.0f);
            }
        }
    }
    __syncthreads();                            // h2 visible

    // ---- density head: relu(h2 @ Wd + bd) (VALU + wave reduce) ----------
    {
        const int row = tid >> 1, q = tid & 1;
        float ps = 0.0f;
        #pragma unroll 8
        for (int j = 0; j < H / 2; ++j)
            ps = fmaf((float)act[row][q * (H / 2) + j], Wd[q * (H / 2) + j], ps);
        ps += __shfl_xor(ps, 1);
        if (q == 0)
            rgbd[(size_t)(p0 + row) * 4 + 3] = fmaxf(ps + bd[0], 0.0f);
    }

    // ---- layer 3: hc = relu([h2,dir] @ W3 + b3), 128x128, K=256 (WMMA) ---
    {
        const _Float16* wbase = w3p + (size_t)lane * 16;
        v8f acc[8];
        #pragma unroll
        for (int j = 0; j < 8; ++j) acc[j] = (v8f)(0.0f);

        v16h afrag = frag_cat(abase + koff, abase + koff + 8);
        v16h bnext = frag_cat(wbase, wbase + 8);
        for (int kt = 0; kt < H / 32; ++kt) {
            const v16h acur = afrag;
            if (kt < H / 32 - 1) {
                const _Float16* ap = abase + (kt + 1) * 32 + koff;
                afrag = frag_cat(ap, ap + 8);
            }
            #pragma unroll
            for (int j = 0; j < 8; ++j) {
                const v16h bcur = bnext;
                const _Float16* wp = wbase + (size_t)(kt * 8 + j + 1) * 512;
                bnext = frag_cat(wp, wp + 8);   // over-reads 1KB past w3p; unused
                acc[j] = __builtin_amdgcn_wmma_f32_16x16x32_f16(
                    false, acur, false, bcur, (short)0, acc[j], false, false);
            }
        }
        __syncthreads();                        // h2 reads (incl. density) done
        #pragma unroll
        for (int j = 0; j < 8; ++j) {
            const int c = j * 16 + (lane & 15);
            const float bias = dcol[c];
            #pragma unroll
            for (int v = 0; v < 8; ++v) {
                const int m = mt * 16 + v + mofs;
                act[m][c] = (_Float16)fmaxf(acc[j][v] + bias, 0.0f);
            }
        }
    }
    __syncthreads();                            // hc visible

    // ---- layer 4: rgb = sigmoid(hc @ W4 + b4), K=128 (VALU) --------------
    for (int it = tid; it < MT * 3; it += 256) {
        const int row = it / 3, ch = it - row * 3;
        float s = 0.0f;
        #pragma unroll 8
        for (int j = 0; j < HC; ++j)
            s = fmaf((float)act[row][j], W4[j * 3 + ch], s);
        s += b4[ch];
        rgbd[(size_t)(p0 + row) * 4 + ch] = 1.0f / (1.0f + __expf(-s));
    }
}

// ---------------------------------------------------------------------------
// Compositing: one wave per ray, 4 samples per lane, shfl-based exclusive scan.
// ---------------------------------------------------------------------------
__global__ __launch_bounds__(256) void nerf_composite(
        const float* __restrict__ rgbd, const float* __restrict__ tv,
        float* __restrict__ out, int Nsamp) {
    const int lane = threadIdx.x & 31;
    const int ray  = blockIdx.x * 8 + (threadIdx.x >> 5);

    float dd[4], pre[4], lsum = 0.0f;
    #pragma unroll
    for (int i = 0; i < 4; ++i) {
        const int s = lane * 4 + i;
        const float diff = (s < Nsamp - 1) ? (tv[s + 1] - tv[s]) : 0.0f;
        const float dens = rgbd[((size_t)ray * Nsamp + s) * 4 + 3];
        dd[i]  = dens * diff;
        pre[i] = lsum;          // lane-local exclusive prefix
        lsum  += dd[i];
    }
    // wave inclusive scan of per-lane sums -> exclusive base
    float v = lsum;
    #pragma unroll
    for (int off = 1; off < 32; off <<= 1) {
        float n = __shfl_up(v, off);
        if (lane >= off) v += n;
    }
    const float base = v - lsum;

    float cr = 0.0f, cg = 0.0f, cb = 0.0f;
    #pragma unroll
    for (int i = 0; i < 4; ++i) {
        const int s = lane * 4 + i;
        const float T = __expf(-(base + pre[i]));
        float a = T * (1.0f - __expf(-dd[i]));
        if (s == 0) a = 0.0f;   // reference skips m=0
        const size_t o = ((size_t)ray * Nsamp + s) * 4;
        cr = fmaf(a, rgbd[o + 0], cr);
        cg = fmaf(a, rgbd[o + 1], cg);
        cb = fmaf(a, rgbd[o + 2], cb);
    }
    #pragma unroll
    for (int off = 16; off >= 1; off >>= 1) {
        cr += __shfl_xor(cr, off);
        cg += __shfl_xor(cg, off);
        cb += __shfl_xor(cb, off);
    }
    if (lane == 0) {
        out[(size_t)ray * 3 + 0] = cr;
        out[(size_t)ray * 3 + 1] = cg;
        out[(size_t)ray * 3 + 2] = cb;
    }
}

// ---------------------------------------------------------------------------
extern "C" void kernel_launch(void* const* d_in, const int* in_sizes, int n_in,
                              void* d_out, int out_size, void* d_ws, size_t ws_size,
                              hipStream_t stream) {
    const float* x  = (const float*)d_in[0];
    const float* d  = (const float*)d_in[1];
    const float* tv = (const float*)d_in[2];
    const float* W1 = (const float*)d_in[3];
    const float* b1 = (const float*)d_in[4];
    const float* W2 = (const float*)d_in[5];
    const float* b2 = (const float*)d_in[6];
    const float* Wd = (const float*)d_in[7];
    const float* bd = (const float*)d_in[8];
    const float* W3 = (const float*)d_in[9];
    const float* b3 = (const float*)d_in[10];
    const float* W4 = (const float*)d_in[11];
    const float* b4 = (const float*)d_in[12];

    const int R = in_sizes[0] / 3;     // 2048 rays (B=1)
    const int N = in_sizes[2];         // 128 samples per ray
    const int points = R * N;

    // workspace layout
    char* ws = (char*)d_ws;
    _Float16* w2p  = (_Float16*)(ws);                       // 128 KiB
    _Float16* w3p  = (_Float16*)(ws + 131072);              // 64 KiB
    float*    rgbd = (float*)(ws + 131072 + 65536);         // points*16 B

    nerf_pack_weights<<<(65536 + 32768 + 255) / 256, 256, 0, stream>>>(
        W2, W3, w2p, w3p);

    nerf_mlp<<<points / MT, 256, 0, stream>>>(
        x, d, tv, W1, b1, b2, Wd, bd, W3, b3, W4, b4, w2p, w3p, rgbd, N);

    nerf_composite<<<R / 8, 256, 0, stream>>>(rgbd, tv, (float*)d_out, N);
}
